// GaussianVectorQuantizer_4724464026026
// MI455X (gfx1250) — matrix operand order repaired and checked by reading, not compiled, not run on hardware
//
#include <hip/hip_runtime.h>
#include <hip/hip_bf16.h>

// ---------------------------------------------------------------------------
// GaussianVectorQuantizer forward, fused for gfx1250 (MI455X).
// B=16, N=1024, D=256, K=16, S=512, TEMP=0.5
//
// d_out layout (flat, return order):
//   zq        : B*N*D           = 4,194,304 f32
//   precision : 1 f32
//   prob      : B*N*S           = 8,388,608 f32
//   log_prob  : B*N*S           = 8,388,608 f32
//
// d_ws layout:
//   books_bf  (K*S*D bf16, row-major (k,s,d))  @ 0        (4 MB)
//   booksT_bf (K*D*S bf16, (k,d,s))            @ 4 MB     (4 MB)
//   b2        (K*S f32)                        @ 8 MB     (32 KB)
//   c_probs   (B*K f32)                        @ 8 MB+32K (1 KB)
// ---------------------------------------------------------------------------

#define BB 16
#define NN 1024
#define DD 256
#define KK 16
#define SS 512
#define TEMP_INV 2.0f

typedef __attribute__((ext_vector_type(16))) __bf16 v16bf;
typedef __attribute__((ext_vector_type(8)))  float  v8f;

// Deterministic per-index gumbel noise (JAX threefry is not reproducible here;
// same inputs -> same noise on every call, as the harness requires).
__device__ __forceinline__ float gumbel_hash(unsigned idx, unsigned salt) {
  unsigned x = idx * 747796405u + salt;
  x ^= x >> 17; x *= 0xED5AD4BBu;
  x ^= x >> 11; x *= 0xAC4C1B51u;
  x ^= x >> 15; x *= 0x31848BABu;
  x ^= x >> 14;
  float u = (float)(x >> 8) * (1.0f / 16777216.0f);
  return -__logf(-__logf(u + 1e-10f) + 1e-10f);
}

// A-fragment for v_wmma_f32_16x16x32_bf16 (16x32, contiguous k order, matched by the B loads):
// lane m<16: M=m, K = {0..15}; lane m+16: M=m, K = {16..31},
// stored as pairs per VGPR.
__device__ __forceinline__ v16bf load_a_frag(const __bf16* base, int lane,
                                             int kbase, int ld) {
  const int m  = lane & 15;
  const int hk = (lane & 16) ? 16 : 0;
  const __bf16* p = base + m * ld + kbase;
  v16bf a;
#pragma unroll
  for (int j = 0; j < 4; ++j) {
    a[2 * j]     = p[hk + 2 * j];
    a[2 * j + 1] = p[hk + 2 * j + 1];
    a[8 + 2 * j]     = p[8 + hk + 2 * j];
    a[8 + 2 * j + 1] = p[8 + hk + 2 * j + 1];
  }
  return a;
}

// --------------------------- prep kernels ----------------------------------

__global__ void prep_cprobs(const float* __restrict__ c_logits,
                            const float* __restrict__ log_param_q,
                            const float* __restrict__ log_param_q_cls,
                            float* __restrict__ c_probs,
                            float* __restrict__ out_scalar) {
  __shared__ float y[BB * KK];
  const int t = threadIdx.x;  // 256 = B*K
  const float pq_cls = 0.5f / fmaxf(1.0f + __expf(log_param_q_cls[0]), 1e-10f);
  const float v = (c_logits[t] * pq_cls + gumbel_hash((unsigned)t, 0x9E3779B9u)) * TEMP_INV;
  y[t] = v;
  __syncthreads();
  const int b = t >> 4;
  float m = -3.4e38f;
#pragma unroll
  for (int k = 0; k < KK; ++k) m = fmaxf(m, y[b * KK + k]);
  float s = 0.f;
#pragma unroll
  for (int k = 0; k < KK; ++k) s += __expf(y[b * KK + k] - m);
  c_probs[t] = __expf(v - m) / s;
  if (t == 0) out_scalar[0] = 0.5f / fmaxf(1.0f + __expf(log_param_q[0]), 1e-10f);
}

__global__ void prep_books(const float* __restrict__ books,
                           __bf16* __restrict__ books_bf,
                           __bf16* __restrict__ booksT,
                           float* __restrict__ b2) {
  const int r = blockIdx.x;    // k*S + s  (8192)
  const int d = threadIdx.x;   // 256
  const int k = r >> 9, s = r & (SS - 1);
  const float v = books[(size_t)r * DD + d];
  const __bf16 bv = (__bf16)v;
  books_bf[(size_t)r * DD + d] = bv;
  booksT[(size_t)(k * DD + d) * SS + s] = bv;
  float sq = v * v;
#pragma unroll
  for (int off = 16; off; off >>= 1) sq += __shfl_xor(sq, off, 32);
  __shared__ float part[8];
  if ((d & 31) == 0) part[d >> 5] = sq;
  __syncthreads();
  if (d == 0) {
    float t = 0.f;
#pragma unroll
    for (int i = 0; i < 8; ++i) t += part[i];
    b2[r] = t;
  }
}

// ----------------------------- main kernel ---------------------------------

__global__ __launch_bounds__(256)
void vq_main(const float* __restrict__ ze,
             const __bf16* __restrict__ books_bf,
             const __bf16* __restrict__ booksT,
             const float* __restrict__ b2,
             const float* __restrict__ c_probs,
             const float* __restrict__ log_param_q,
             float* __restrict__ zq_out,
             float* __restrict__ prob_out,
             float* __restrict__ logp_out) {
  __shared__ __bf16 ze_bf[16 * DD];       //  8 KB
  __shared__ float  z2[16];
  __shared__ float  cp[KK];
  __shared__ float  logits_acc[16 * SS];  // 32 KB
  __shared__ float  Lg[16 * SS];          // 32 KB
  __shared__ __bf16 enc_bf[16 * SS];      // 16 KB

  const int b    = blockIdx.y;
  const int n0   = blockIdx.x * 16;
  const int t    = threadIdx.x;
  const int w    = t >> 5;
  const int lane = t & 31;
  const int hi   = (lane & 16) ? 1 : 0;
  const float precision_q = 0.5f / fmaxf(1.0f + __expf(log_param_q[0]), 1e-10f);

  if (t < 16) { z2[t] = 0.f; cp[t] = c_probs[b * KK + t]; }
  for (int i = t; i < 16 * SS; i += 256) logits_acc[i] = 0.f;
  __syncthreads();

  // Load ze tile (16 x 256), convert to bf16 in LDS, compute row sq-norms.
#pragma unroll 4
  for (int i = 0; i < 16; ++i) {
    const float v = ze[(size_t)(b * NN + n0 + i) * DD + t];
    ze_bf[i * DD + t] = (__bf16)v;
    float sq = v * v;
#pragma unroll
    for (int off = 16; off; off >>= 1) sq += __shfl_xor(sq, off, 32);
    if (lane == 0) atomicAdd(&z2[i], sq);
  }
  __syncthreads();

  // Persistent zq accumulators: wave w owns d in [32w, 32w+32).
  v8f accq0 = {};
  v8f accq1 = {};
  const int d0 = w * 32;

  for (int k = 0; k < KK; ++k) {
    const float cpk = cp[k];
    // ---- Phase 1: 16x512 logit block for cluster k (4 s-tiles per wave) ----
    for (int tt = 0; tt < 4; ++tt) {
      const int sbase = (w + tt * 8) * 16;
      v8f acc = {};
      const __bf16* bp =
          books_bf + ((size_t)(k * SS + sbase + (lane & 15)) * DD + hi * 16);
#pragma unroll
      for (int dc = 0; dc < 8; ++dc) {
        v16bf a  = load_a_frag(ze_bf, lane, dc * 32, DD);
        v16bf bb = *(const v16bf*)(bp + dc * 32);
        acc = __builtin_amdgcn_wmma_f32_16x16x32_bf16(
            false, a, false, bb, (short)0, acc, false, false);
      }
      const int ss = sbase + (lane & 15);
      const float b2v = b2[k * SS + ss];
#pragma unroll
      for (int r = 0; r < 8; ++r) {
        const int mm = r + hi * 8;
        const float l = (2.f * acc[r] - z2[mm] - b2v) * precision_q;
        logits_acc[mm * SS + ss] += cpk * l;  // each (mm,ss) owned by one lane
        const unsigned idx =
            (((unsigned)(b * NN + n0 + mm) * KK + (unsigned)k) * SS + (unsigned)ss);
        Lg[mm * SS + ss] = (l + gumbel_hash(idx, 0x85EBCA6Bu)) * TEMP_INV;
      }
    }
    __syncthreads();

    // ---- Phase 2: gumbel-softmax rows (2 rows per wave), scale by cp[k] ----
#pragma unroll
    for (int rr = 0; rr < 2; ++rr) {
      const int m = w * 2 + rr;
      float vals[16];
      float vmax = -3.4e38f;
#pragma unroll
      for (int i = 0; i < 16; ++i) {
        vals[i] = Lg[m * SS + lane + 32 * i];
        vmax = fmaxf(vmax, vals[i]);
      }
#pragma unroll
      for (int off = 16; off; off >>= 1) vmax = fmaxf(vmax, __shfl_xor(vmax, off, 32));
      float sum = 0.f;
#pragma unroll
      for (int i = 0; i < 16; ++i) { vals[i] = __expf(vals[i] - vmax); sum += vals[i]; }
#pragma unroll
      for (int off = 16; off; off >>= 1) sum += __shfl_xor(sum, off, 32);
      const float scale = cpk / sum;
#pragma unroll
      for (int i = 0; i < 16; ++i)
        enc_bf[m * SS + lane + 32 * i] = (__bf16)(vals[i] * scale);
    }
    __syncthreads();

    // ---- Phase 3: zq += (cp*enc) @ books[k]  (wave-private 32-wide D slice) --
    const __bf16* bt0 =
        booksT + ((size_t)(k * DD + d0 + (lane & 15)) * SS + hi * 16);
    const __bf16* bt1 = bt0 + (size_t)16 * SS;
#pragma unroll
    for (int sc = 0; sc < 16; ++sc) {
      v16bf a  = load_a_frag(enc_bf, lane, sc * 32, SS);
      v16bf b0 = *(const v16bf*)(bt0 + sc * 32);
      accq0 = __builtin_amdgcn_wmma_f32_16x16x32_bf16(
          false, a, false, b0, (short)0, accq0, false, false);
      v16bf b1 = *(const v16bf*)(bt1 + sc * 32);
      accq1 = __builtin_amdgcn_wmma_f32_16x16x32_bf16(
          false, a, false, b1, (short)0, accq1, false, false);
    }
    __syncthreads();
  }

  // ---- zq writeback ----
#pragma unroll
  for (int r = 0; r < 8; ++r) {
    const int mm = r + hi * 8;
    const size_t base = (size_t)(b * NN + n0 + mm) * DD + d0 + (lane & 15);
    zq_out[base]      = accq0[r];
    zq_out[base + 16] = accq1[r];
  }

  // ---- prob / log_prob over S (2 rows per wave) ----
#pragma unroll
  for (int rr = 0; rr < 2; ++rr) {
    const int m = w * 2 + rr;
    float vals[16], ev[16];
    float vmax = -3.4e38f;
#pragma unroll
    for (int i = 0; i < 16; ++i) {
      vals[i] = logits_acc[m * SS + lane + 32 * i];
      vmax = fmaxf(vmax, vals[i]);
    }
#pragma unroll
    for (int off = 16; off; off >>= 1) vmax = fmaxf(vmax, __shfl_xor(vmax, off, 32));
    float sum = 0.f;
#pragma unroll
    for (int i = 0; i < 16; ++i) { ev[i] = __expf(vals[i] - vmax); sum += ev[i]; }
#pragma unroll
    for (int off = 16; off; off >>= 1) sum += __shfl_xor(sum, off, 32);
    const float inv = 1.0f / sum;
    const float lse = __logf(sum);
    const size_t base = (size_t)(b * NN + n0 + m) * SS;
#pragma unroll
    for (int i = 0; i < 16; ++i) {
      const int s = lane + 32 * i;
      prob_out[base + s] = ev[i] * inv;
      logp_out[base + s] = vals[i] - vmax - lse;
    }
  }
}

// ------------------------------- launcher ----------------------------------

extern "C" void kernel_launch(void* const* d_in, const int* in_sizes, int n_in,
                              void* d_out, int out_size, void* d_ws, size_t ws_size,
                              hipStream_t stream) {
  (void)in_sizes; (void)n_in; (void)out_size; (void)ws_size;
  const float* ze       = (const float*)d_in[0];
  const float* c_logits = (const float*)d_in[1];
  const float* books    = (const float*)d_in[2];
  const float* lpq      = (const float*)d_in[3];
  const float* lpq_cls  = (const float*)d_in[4];
  // d_in[5] = is_train (training path hardcoded)

  char* ws = (char*)d_ws;
  __bf16* books_bf = (__bf16*)ws;                                   // 4 MB
  __bf16* booksT   = (__bf16*)(ws + ((size_t)4 << 20));             // 4 MB
  float*  b2       = (float*)(ws + ((size_t)8 << 20));              // 32 KB
  float*  c_probs  = (float*)(ws + ((size_t)8 << 20) + 32 * 1024);  // 1 KB

  float* out    = (float*)d_out;
  float* zq     = out;
  float* scalar = out + (size_t)BB * NN * DD;
  float* prob   = scalar + 1;
  float* logp   = prob + (size_t)BB * NN * SS;

  prep_cprobs<<<1, BB * KK, 0, stream>>>(c_logits, lpq, lpq_cls, c_probs, scalar);
  prep_books<<<KK * SS, DD, 0, stream>>>(books, books_bf, booksT, b2);
  vq_main<<<dim3(NN / 16, BB), 256, 0, stream>>>(ze, books_bf, booksT, b2,
                                                 c_probs, lpq, zq, prob, logp);
}
